// GBlockLSTMCell_50087908606418
// MI455X (gfx1250) — compile-verified, matrix operand order checked by reading, not run on hardware
//
#include <hip/hip_runtime.h>
#include <hip/hip_bf16.h>

// ---------------------------------------------------------------------------
// GBlockLSTMCell fused kernel for gfx1250 (MI455X)
//   gates = x_t @ Win^T  +  [A kron I + B kron (J-I)] applied to h_prev
//   i,f,o = sigmoid ; g = tanh ; c = f*c_prev + i*g ; h = o*tanh(c)
// bf16 WMMA (v_wmma_f32_16x16x32_bf16) for the dense K=2048 GEMM and the
// K=128 block-structured term; fully fused 4-gate LSTM epilogue.
// Addressing: uniform SGPR kernel-arg bases + 32-bit per-lane offsets so the
// backend emits SADDR-form global_load_b128 with immediate K offsets instead
// of carrying 64-bit per-lane address pairs through the loop.
// ---------------------------------------------------------------------------

typedef __bf16 bf16_t;
typedef __attribute__((ext_vector_type(8)))  bf16_t v8bf;
typedef __attribute__((ext_vector_type(16))) bf16_t v16bf;
typedef __attribute__((ext_vector_type(8)))  float  v8f;

#define BSZ   1024
#define INDIM 2048
#define HDIM  2048
#define HBLK  128
#define NGATE 4

// ---------------- f32 -> bf16 (RNE) conversion -----------------------------
__device__ __forceinline__ unsigned short f2bf(float f) {
    union { float f; unsigned int u; } x; x.f = f;
    unsigned int u = x.u;
    u += 0x7fffu + ((u >> 16) & 1u);     // round-to-nearest-even
    return (unsigned short)(u >> 16);
}

__global__ void cvt_f32_bf16_kernel(const float* __restrict__ in,
                                    unsigned short* __restrict__ out, int n) {
    int i = blockIdx.x * blockDim.x + threadIdx.x;
    if (i < n) out[i] = f2bf(in[i]);
}

// ---------------- build u = hp, v = (sum_k hp) - hp  (bf16) ----------------
__global__ void build_uv_kernel(const float* __restrict__ h_prev,
                                unsigned short* __restrict__ u,
                                unsigned short* __restrict__ v) {
    int idx = blockIdx.x * blockDim.x + threadIdx.x;   // 1024*128 threads
    int b = idx >> 7;
    int j = idx & (HBLK - 1);
    float hp[HDIM / HBLK];
    float s = 0.f;
#pragma unroll
    for (int k = 0; k < HDIM / HBLK; ++k) {
        hp[k] = h_prev[(size_t)b * HDIM + k * HBLK + j];
        s += hp[k];
    }
#pragma unroll
    for (int k = 0; k < HDIM / HBLK; ++k) {
        size_t o = (size_t)b * HDIM + k * HBLK + j;
        u[o] = f2bf(hp[k]);
        v[o] = f2bf(s - hp[k]);
    }
}

// ---------------- WMMA helpers ---------------------------------------------
// 16-bit operand layout: per lane, two contiguous 8-element (16 B) K-chunks.
// Base pointer is a uniform kernel arg; `off` is a 32-bit element offset so
// the backend can use saddr-form loads with immediate offsets.
__device__ __forceinline__ v16bf load_frag(const bf16_t* __restrict__ base,
                                           unsigned off) {
    v8bf lo = *(const v8bf*)(base + off);
    v8bf hi = *(const v8bf*)(base + off + 16);
    return __builtin_shufflevector(lo, hi, 0, 1, 2, 3, 4, 5, 6, 7,
                                          8, 9, 10, 11, 12, 13, 14, 15);
}

__device__ __forceinline__ v8f wmma_bf16(v16bf a, v16bf b, v8f c) {
    return __builtin_amdgcn_wmma_f32_16x16x32_bf16(
        false, a, false, b, (short)0, c, false, false);
}

__device__ __forceinline__ float sigmoidf_fast(float x) {
    return 1.0f / (1.0f + __expf(-x));
}

__device__ __forceinline__ float tanhf_fast(float x) {
    x = fminf(fmaxf(x, -30.0f), 30.0f);
    float e = __expf(2.0f * x);
    return (e - 1.0f) / (e + 1.0f);
}

// ---------------- fused GEMM + block term + LSTM epilogue ------------------
// Workgroup: 256 threads = 8 waves in a 2x4 grid.
//   WG tile : 64 batch rows x 128 h-columns (one full H-block) x 4 gates
//   Wave    : 32 x 32 per gate = 2x2 WMMA 16x16 tiles, 4 gates
__global__ __launch_bounds__(256)
void fused_gblock_lstm_kernel(const bf16_t* __restrict__ X,    // [B, IN]
                              const bf16_t* __restrict__ W0,   // [H, IN] gate 0
                              const bf16_t* __restrict__ W1,   // gate 1
                              const bf16_t* __restrict__ W2,   // gate 2
                              const bf16_t* __restrict__ W3,   // gate 3
                              const bf16_t* __restrict__ U,    // [B, H]
                              const bf16_t* __restrict__ V,    // [B, H]
                              const bf16_t* __restrict__ Ab,   // [4,128,128]
                              const bf16_t* __restrict__ Bb,   // [4,128,128]
                              const float*  __restrict__ c_prev,
                              float* __restrict__ h_out,
                              float* __restrict__ c_out) {
    const int lane = threadIdx.x & 31;
    const int wave = threadIdx.x >> 5;
    const int wr   = wave >> 2;          // 0..1
    const int wc   = wave & 3;           // 0..3
    const int m16  = lane & 15;
    const int half = lane >> 4;

    const int rowBase  = blockIdx.y * 64 + wr * 32;        // batch row
    const int colBase  = blockIdx.x * 128 + wc * 32;       // h column (0..2047)
    const int blk      = blockIdx.x;                       // H-block index
    const int jcolBase = wc * 32;                          // col within block

    const bf16_t* __restrict__ Wg[NGATE] = { W0, W1, W2, W3 };

    v8f acc[NGATE][2][2];
#pragma unroll
    for (int g = 0; g < NGATE; ++g)
#pragma unroll
        for (int a = 0; a < 2; ++a)
#pragma unroll
            for (int b = 0; b < 2; ++b)
                acc[g][a][b] = (v8f){};

    // 32-bit per-lane element offsets (uniform bases stay in SGPRs).
    //   tm/tn deltas (16*2048 elts = 64 KiB) and the whole K sweep (4 KiB)
    //   are reachable via the 24-bit instruction immediate.
    const unsigned offX = (unsigned)(rowBase + m16) * INDIM + half * 8u;
    const unsigned offB = (unsigned)(colBase + m16) * INDIM + half * 8u;

    // ---- Phase 1: dense GEMM, K = 2048 ------------------------------------
    for (unsigned k = 0; k < INDIM; k += 32) {
        v16bf a0 = load_frag(X, offX + k);
        v16bf a1 = load_frag(X, offX + 16u * INDIM + k);
#pragma unroll
        for (int g = 0; g < NGATE; ++g)
#pragma unroll
            for (int tn = 0; tn < 2; ++tn) {
                v16bf b = load_frag(Wg[g], offB + (unsigned)tn * 16u * INDIM + k);
                acc[g][0][tn] = wmma_bf16(a0, b, acc[g][0][tn]);
                acc[g][1][tn] = wmma_bf16(a1, b, acc[g][1][tn]);
            }
    }

    // ---- Phase 2/3: + u @ A[g]^T + v @ B[g]^T, K = 128 each ---------------
    // Gate stride in Ab/Bb = 128*128 elts = 32 KiB -> immediate-reachable.
    const unsigned offUV = (unsigned)(rowBase + m16) * HDIM + (unsigned)blk * HBLK + half * 8u;
    const unsigned offAB = (unsigned)(jcolBase + m16) * HBLK + half * 8u;

#pragma unroll
    for (int k = 0; k < HBLK; k += 32) {
        v16bf a0 = load_frag(U, offUV + k);
        v16bf a1 = load_frag(U, offUV + 16u * HDIM + k);
#pragma unroll
        for (int g = 0; g < NGATE; ++g)
#pragma unroll
            for (int tn = 0; tn < 2; ++tn) {
                v16bf b = load_frag(Ab, offAB + (unsigned)(g * HBLK * HBLK + tn * 16 * HBLK) + k);
                acc[g][0][tn] = wmma_bf16(a0, b, acc[g][0][tn]);
                acc[g][1][tn] = wmma_bf16(a1, b, acc[g][1][tn]);
            }
    }

#pragma unroll
    for (int k = 0; k < HBLK; k += 32) {
        v16bf a0 = load_frag(V, offUV + k);
        v16bf a1 = load_frag(V, offUV + 16u * HDIM + k);
#pragma unroll
        for (int g = 0; g < NGATE; ++g)
#pragma unroll
            for (int tn = 0; tn < 2; ++tn) {
                v16bf b = load_frag(Bb, offAB + (unsigned)(g * HBLK * HBLK + tn * 16 * HBLK) + k);
                acc[g][0][tn] = wmma_bf16(a0, b, acc[g][0][tn]);
                acc[g][1][tn] = wmma_bf16(a1, b, acc[g][1][tn]);
            }
    }

    // ---- Fused LSTM epilogue ----------------------------------------------
    // C/D 16x16 f32 layout: lane = N (mod 16), VGPR r -> M = r + 8*half.
#pragma unroll
    for (int tm = 0; tm < 2; ++tm)
#pragma unroll
        for (int tn = 0; tn < 2; ++tn) {
            const int col = colBase + tn * 16 + m16;
#pragma unroll
            for (int r = 0; r < 8; ++r) {
                const int row = rowBase + tm * 16 + half * 8 + r;
                const unsigned o = (unsigned)row * HDIM + col;
                float iv = acc[0][tm][tn][r];
                float fv = acc[1][tm][tn][r];
                float gv = acc[2][tm][tn][r];
                float ov = acc[3][tm][tn][r];
                float ig = sigmoidf_fast(iv);
                float fg = sigmoidf_fast(fv);
                float og = sigmoidf_fast(ov);
                float gt = tanhf_fast(gv);
                float cn = fg * c_prev[o] + ig * gt;
                float hn = og * tanhf_fast(cn);
                c_out[o] = cn;
                h_out[o] = hn;
            }
        }
}

// ---------------------------------------------------------------------------
extern "C" void kernel_launch(void* const* d_in, const int* in_sizes, int n_in,
                              void* d_out, int out_size, void* d_ws, size_t ws_size,
                              hipStream_t stream) {
    const float* x_t    = (const float*)d_in[0];   // [1024, 2048]
    const float* h_prev = (const float*)d_in[1];   // [1024, 2048]
    const float* c_prev = (const float*)d_in[2];   // [1024, 2048]
    const float* Win    = (const float*)d_in[3];   // [8192, 2048]
    const float* A      = (const float*)d_in[4];   // [4, 128, 128]
    const float* Bm     = (const float*)d_in[5];   // [4, 128, 128]

    float* h_out = (float*)d_out;                          // tuple elem 0
    float* c_out = h_out + (size_t)BSZ * HDIM;             // tuple elem 1

    // Workspace layout (bf16 stored as ushort): ~46.4 MB total
    unsigned short* ws  = (unsigned short*)d_ws;
    unsigned short* Wbf = ws;                                  // 8192*2048
    unsigned short* Xbf = Wbf + (size_t)NGATE * HDIM * INDIM;  // 1024*2048
    unsigned short* Ubf = Xbf + (size_t)BSZ * INDIM;
    unsigned short* Vbf = Ubf + (size_t)BSZ * HDIM;
    unsigned short* Abf = Vbf + (size_t)BSZ * HDIM;            // 4*128*128
    unsigned short* Bbf = Abf + (size_t)NGATE * HBLK * HBLK;

    const int nW  = NGATE * HDIM * INDIM;   // 16,777,216
    const int nX  = BSZ * INDIM;            // 2,097,152
    const int nAB = NGATE * HBLK * HBLK;    // 65,536

    cvt_f32_bf16_kernel<<<(nW  + 255) / 256, 256, 0, stream>>>(Win, Wbf, nW);
    cvt_f32_bf16_kernel<<<(nX  + 255) / 256, 256, 0, stream>>>(x_t, Xbf, nX);
    cvt_f32_bf16_kernel<<<(nAB + 255) / 256, 256, 0, stream>>>(A,  Abf, nAB);
    cvt_f32_bf16_kernel<<<(nAB + 255) / 256, 256, 0, stream>>>(Bm, Bbf, nAB);
    build_uv_kernel<<<(BSZ * HBLK) / 256, 256, 0, stream>>>(h_prev, Ubf, Vbf);

    const bf16_t* Wb = (const bf16_t*)Wbf;
    dim3 grid(HDIM / 128, BSZ / 64);        // 16 x 16 workgroups
    fused_gblock_lstm_kernel<<<grid, 256, 0, stream>>>(
        (const bf16_t*)Xbf,
        Wb + 0 * (size_t)HDIM * INDIM,
        Wb + 1 * (size_t)HDIM * INDIM,
        Wb + 2 * (size_t)HDIM * INDIM,
        Wb + 3 * (size_t)HDIM * INDIM,
        (const bf16_t*)Ubf, (const bf16_t*)Vbf,
        (const bf16_t*)Abf, (const bf16_t*)Bbf,
        c_prev, h_out, c_out);
}
